// RNN_21139829031255
// MI455X (gfx1250) — matrix operand order repaired and checked by reading, not compiled, not run on hardware
//
#include <hip/hip_runtime.h>
#include <stdint.h>

typedef __attribute__((ext_vector_type(16))) __bf16 v16bf;
typedef __attribute__((ext_vector_type(8)))  float  v8f;

#define B_   32
#define T_   512
#define D_   1024
#define H_   1024
#define G4_  4096   // 4*H

__device__ __forceinline__ __bf16 f2bf(float f) {
    uint32_t u = __builtin_bit_cast(uint32_t, f);
    u += 0x7FFFu + ((u >> 16) & 1u);        // round-to-nearest-even
    uint16_t h = (uint16_t)(u >> 16);
    return __builtin_bit_cast(__bf16, h);
}

__device__ __forceinline__ float sigmoidf_(float x) {
    return 1.0f / (1.0f + __expf(-x));
}

union Frag { v16bf v; uint4 u[2]; };

// ---------------------------------------------------------------- conversions
__global__ void cvt_f32_bf16_kernel(const float* __restrict__ src,
                                    __bf16* __restrict__ dst, int n) {
    int i = blockIdx.x * blockDim.x + threadIdx.x;
    if (i < n) dst[i] = f2bf(src[i]);
}

// ---------------------------------------------------------------- input-proj GEMM
// gx[t][b][n] = sum_k X[b*T+t][k] * W[n][k] + bias[n]
// X: [M,K] bf16 row-major (row m = b*T_ + t), W: [G4,K] bf16 row-major.
// One wave computes a 16x64 strip; K loop is manually double-buffered so the
// next 32-deep fragment set is in flight while the current WMMAs execute.
__global__ __launch_bounds__(128)
void gemm_gx_kernel(const __bf16* __restrict__ X,
                    const __bf16* __restrict__ W,
                    const float*  __restrict__ bias,
                    float* __restrict__ gx,         // [T_, B_, G4_]
                    int M, int K)                   // K % 64 == 0
{
    const int lane = threadIdx.x & 31;
    const int wave = blockIdx.x * (blockDim.x >> 5) + (threadIdx.x >> 5);
    const int nStrips = G4_ >> 6;                 // 64-col strips
    const int mTiles  = M >> 4;
    if (wave >= mTiles * nStrips) return;
    const int mt = wave / nStrips;
    const int ns = wave % nStrips;

    const int hiA  = (lane >> 4) << 3;            // 0 or 8  (A K-offset)
    const int hiB  = (lane >> 4) << 3;            // 0 or 8  (B K-offset)
    const __bf16* aBase = X + (size_t)((mt << 4) + (lane & 15)) * K + hiA;
    const __bf16* bBase = W + (size_t)((ns << 6) + (lane & 15)) * K + hiB;

    auto loadA = [&](int k0, Frag& a) {
        const uint4* p = (const uint4*)(aBase + k0);
        a.u[0] = p[0];                            // K = k0+hiA .. +7
        a.u[1] = p[2];                            // K = k0+hiA+16 .. +23
    };
    auto loadB4 = [&](int k0, Frag* b) {
#pragma unroll
        for (int j = 0; j < 4; ++j) {
            const uint4* p = (const uint4*)(bBase + (size_t)(j << 4) * K + k0);
            b[j].u[0] = p[0];                     // K = k0+hiB .. +7
            b[j].u[1] = p[2];                     // K = k0+hiB+16 .. +23
        }
    };

    v8f acc[4] = { {}, {}, {}, {} };
    Frag a0, a1, b0[4], b1[4];
    loadA(0, a0);
    loadB4(0, b0);
    for (int k0 = 0; k0 < K; k0 += 64) {
        loadA(k0 + 32, a1);                       // prefetch next half-step
        loadB4(k0 + 32, b1);
#pragma unroll
        for (int j = 0; j < 4; ++j)
            acc[j] = __builtin_amdgcn_wmma_f32_16x16x32_bf16(
                false, a0.v, false, b0[j].v, (short)0, acc[j], false, false);
        if (k0 + 64 < K) {
            loadA(k0 + 64, a0);
            loadB4(k0 + 64, b0);
        }
#pragma unroll
        for (int j = 0; j < 4; ++j)
            acc[j] = __builtin_amdgcn_wmma_f32_16x16x32_bf16(
                false, a1.v, false, b1[j].v, (short)0, acc[j], false, false);
    }

    const int nBase = ns << 6;
#pragma unroll
    for (int j = 0; j < 4; ++j) {
        const int n  = nBase + (j << 4) + (lane & 15);
        const float bv = bias[n];
#pragma unroll
        for (int r = 0; r < 8; ++r) {
            const int row = (mt << 4) + r + hiA;  // row = b*T_ + t
            const int bi  = row / T_;
            const int ti  = row % T_;
            gx[((size_t)ti * B_ + bi) * G4_ + n] = acc[j][r] + bv;
        }
    }
}

// ---------------------------------------------------------------- LSTM timestep
// gates = gx[t] + h @ W_hh^T.
// One 256-thread block (8 waves) owns one 16x16 (b,n) tile.
// Wave w = kh*4 + g computes gate group g over K-half kh (512-deep slice,
// 16 WMMAs). Partials meet in LDS; after the barrier, wave w handles
// accumulator row r=w of the fused activation / cell-update epilogue.
__global__ __launch_bounds__(256)
void lstm_step_kernel(const float*  __restrict__ gx_t,    // [B_, G4_]
                      const __bf16* __restrict__ Whh,     // [G4_, H_]
                      const __bf16* __restrict__ hbf_in,  // [B_, H_]
                      float* __restrict__ c_state,        // [B_, H_]
                      float* __restrict__ h_state,        // [B_, H_]
                      __bf16* __restrict__ hbf_out,       // [B_, H_]
                      const long long* __restrict__ lengths,
                      int t,
                      float*  __restrict__ out_f32,       // [B_,T_,H_] or null
                      __bf16* __restrict__ out_bf)        // [B_,T_,H_] or null
{
    __shared__ float red[8 * 8 * 32];              // [wave][reg][lane], 8 KB

    const int lane = threadIdx.x & 31;
    const int wv   = threadIdx.x >> 5;             // 0..7
    const int g    = wv & 3;                       // gate group
    const int kh   = wv >> 2;                      // K half
    const int mt   = blockIdx.x >> 6;              // B_=32 -> 2 m-tiles
    const int nt   = blockIdx.x & 63;              // H_/16 = 64 n-tiles

    const int hiA  = (lane >> 4) << 3;
    const int hiB  = (lane >> 4) << 3;
    const int rowA = (mt << 4) + (lane & 15);      // batch row
    const int colN = (nt << 4) + (lane & 15);      // hidden col

    const __bf16* aBase = hbf_in + (size_t)rowA * H_ + hiA;
    const __bf16* bBase = Whh + (size_t)(g * H_ + colN) * H_ + hiB;

    const int kBeg = kh * (H_ / 2);
    const int kEnd = kBeg + (H_ / 2);

    v8f acc = {};
    for (int k0 = kBeg; k0 < kEnd; k0 += 32) {
        Frag a, b;
        const uint4* pa = (const uint4*)(aBase + k0);
        a.u[0] = pa[0];
        a.u[1] = pa[2];
        const uint4* pb = (const uint4*)(bBase + k0);
        b.u[0] = pb[0];
        b.u[1] = pb[2];
        __builtin_prefetch((const void*)(bBase + k0 + 128), 0, 1); // 4 iters ahead
        acc = __builtin_amdgcn_wmma_f32_16x16x32_bf16(
            false, a.v, false, b.v, (short)0, acc, false, false);
    }

#pragma unroll
    for (int r = 0; r < 8; ++r)
        red[(wv * 8 + r) * 32 + lane] = acc[r];
    __syncthreads();

    // epilogue: wave wv owns accumulator row r = wv
    {
        const int r  = wv;
        const int b  = (mt << 4) + r + hiA;
        const int n  = colN;
        const float* gxr = gx_t + (size_t)b * G4_;
        const int e = r * 32 + lane;
        const float vf  = red[(0 * 8) * 32 + e] + red[(4 * 8) * 32 + e];
        const float vi  = red[(1 * 8) * 32 + e] + red[(5 * 8) * 32 + e];
        const float vo  = red[(2 * 8) * 32 + e] + red[(6 * 8) * 32 + e];
        const float vcg = red[(3 * 8) * 32 + e] + red[(7 * 8) * 32 + e];
        const float f  = sigmoidf_(vf  + gxr[n]);
        const float i  = sigmoidf_(vi  + gxr[H_  + n]);
        const float o  = sigmoidf_(vo  + gxr[2*H_ + n]);
        const float cg = tanhf    (vcg + gxr[3*H_ + n]);
        const size_t si = (size_t)b * H_ + n;
        const float co = c_state[si];
        const float ho = h_state[si];
        float cn = f * co + i * cg;
        float hn = o * tanhf(cn);
        const bool m = ((long long)t < lengths[b]);
        cn = m ? cn : co;
        hn = m ? hn : ho;
        c_state[si] = cn;
        h_state[si] = hn;
        hbf_out[si] = f2bf(hn);
        if (out_f32) out_f32[((size_t)b * T_ + t) * H_ + n] = hn;
        if (out_bf)  out_bf [((size_t)b * T_ + t) * H_ + n] = f2bf(hn);
    }
}

// ---------------------------------------------------------------- state init / finalize
__global__ void init_state_kernel(float* c, float* h, __bf16* hbf, int n) {
    int i = blockIdx.x * blockDim.x + threadIdx.x;
    if (i < n) { c[i] = 0.f; h[i] = 0.f; hbf[i] = f2bf(0.f); }
}

__global__ void finalize_kernel(const float* __restrict__ c0,
                                const float* __restrict__ h0,
                                const float* __restrict__ c1,
                                const float* __restrict__ h1,
                                float* __restrict__ out) {
    int i = blockIdx.x * blockDim.x + threadIdx.x;
    const int n = B_ * H_;
    if (i < n) {
        float* cbase = out + (size_t)B_ * T_ * H_;
        float* hbase = cbase + 2 * (size_t)n;
        cbase[i]     = c0[i];
        cbase[n + i] = c1[i];
        hbase[i]     = h0[i];
        hbase[n + i] = h1[i];
    }
}

// ---------------------------------------------------------------- launcher
extern "C" void kernel_launch(void* const* d_in, const int* in_sizes, int n_in,
                              void* d_out, int out_size, void* d_ws, size_t ws_size,
                              hipStream_t stream) {
    const float*      x       = (const float*)d_in[0];
    const long long*  lengths = (const long long*)d_in[1];
    const float*      w_ih_0  = (const float*)d_in[2];
    const float*      w_hh_0  = (const float*)d_in[3];
    const float*      b_0     = (const float*)d_in[4];
    const float*      w_ih_1  = (const float*)d_in[5];
    const float*      w_hh_1  = (const float*)d_in[6];
    const float*      b_1     = (const float*)d_in[7];
    float*            out     = (float*)d_out;

    char* ws = (char*)d_ws;
    size_t off = 0;
    auto alloc = [&](size_t bytes) -> void* {
        void* p = ws + off;
        off += (bytes + 255) & ~(size_t)255;
        return p;
    };
    __bf16* Wih0 = (__bf16*)alloc((size_t)G4_ * D_ * 2);
    __bf16* Whh0 = (__bf16*)alloc((size_t)G4_ * H_ * 2);
    __bf16* Wih1 = (__bf16*)alloc((size_t)G4_ * H_ * 2);
    __bf16* Whh1 = (__bf16*)alloc((size_t)G4_ * H_ * 2);
    __bf16* xbf  = (__bf16*)alloc((size_t)B_ * T_ * D_ * 2);
    __bf16* x1bf = (__bf16*)alloc((size_t)B_ * T_ * H_ * 2);
    float*  gx   = (float*) alloc((size_t)T_ * B_ * G4_ * 4);
    float*  c0s  = (float*) alloc((size_t)B_ * H_ * 4);
    float*  h0s  = (float*) alloc((size_t)B_ * H_ * 4);
    float*  c1s  = (float*) alloc((size_t)B_ * H_ * 4);
    float*  h1s  = (float*) alloc((size_t)B_ * H_ * 4);
    __bf16* hA   = (__bf16*)alloc((size_t)B_ * H_ * 2);
    __bf16* hB   = (__bf16*)alloc((size_t)B_ * H_ * 2);

    // --- precision conversions (one time) ---
    const int nX = B_ * T_ * D_;
    cvt_f32_bf16_kernel<<<(nX + 255) / 256, 256, 0, stream>>>(x, xbf, nX);
    const int nW0 = G4_ * D_;
    const int nW  = G4_ * H_;
    cvt_f32_bf16_kernel<<<(nW0 + 255) / 256, 256, 0, stream>>>(w_ih_0, Wih0, nW0);
    cvt_f32_bf16_kernel<<<(nW  + 255) / 256, 256, 0, stream>>>(w_hh_0, Whh0, nW);
    cvt_f32_bf16_kernel<<<(nW  + 255) / 256, 256, 0, stream>>>(w_ih_1, Wih1, nW);
    cvt_f32_bf16_kernel<<<(nW  + 255) / 256, 256, 0, stream>>>(w_hh_1, Whh1, nW);

    const int M      = B_ * T_;
    const int strips = (M / 16) * (G4_ / 64);          // waves needed
    const int gemmBlocks = (strips + 3) / 4;           // 4 waves / 128-thread block
    const int nS = B_ * H_;
    const int stepBlocks = 2 * (H_ / 16);              // one block per 16x16 tile

    // ================= layer 0 =================
    gemm_gx_kernel<<<gemmBlocks, 128, 0, stream>>>(xbf, Wih0, b_0, gx, M, D_);
    init_state_kernel<<<(nS + 255) / 256, 256, 0, stream>>>(c0s, h0s, hA, nS);
    for (int t = 0; t < T_; ++t) {
        const __bf16* hin  = (t & 1) ? hB : hA;
        __bf16*       hout = (t & 1) ? hA : hB;
        lstm_step_kernel<<<stepBlocks, 256, 0, stream>>>(
            gx + (size_t)t * B_ * G4_, Whh0, hin, c0s, h0s, hout,
            lengths, t, nullptr, x1bf);
    }

    // ================= layer 1 =================
    gemm_gx_kernel<<<gemmBlocks, 128, 0, stream>>>(x1bf, Wih1, b_1, gx, M, H_);
    init_state_kernel<<<(nS + 255) / 256, 256, 0, stream>>>(c1s, h1s, hA, nS);
    for (int t = 0; t < T_; ++t) {
        const __bf16* hin  = (t & 1) ? hB : hA;
        __bf16*       hout = (t & 1) ? hA : hB;
        lstm_step_kernel<<<stepBlocks, 256, 0, stream>>>(
            gx + (size_t)t * B_ * G4_, Whh1, hin, c1s, h1s, hout,
            lengths, t, out, nullptr);
    }

    // ================= (c,h) tail =================
    finalize_kernel<<<(nS + 255) / 256, 256, 0, stream>>>(c0s, h0s, c1s, h1s, out);
}
